// GraphAttnBias_11897059410767
// MI455X (gfx1250) — matrix operand order repaired and checked by reading, not compiled, not run on hardware
//
#include <hip/hip_runtime.h>
#include <hip/hip_bf16.h>

#define G_     32
#define N_     128
#define H_     32
#define D_     5
#define NEDGE  65536
#define NPATH  1048576
#define N1     129

typedef __attribute__((ext_vector_type(16))) _Float16 v16h;
typedef __attribute__((ext_vector_type(8)))  _Float16 v8h;
typedef __attribute__((ext_vector_type(8)))  float    v8f;

// workspace layout (bytes)
#define OFF_BPACK 0u                      // 2*5*32*16 f16 = 10240 B (pad to 16 KiB)
#define OFF_EA16  16384u                  // 65536*32 f16 = 4 MiB
#define OFF_ADJ   (16384u + 4194304u)     // 524288 i32
#define OFF_PSLOT (OFF_ADJ + 2097152u)    // 2621440 i32
// total ~16.8 MiB

// ---- init index maps to -1 -------------------------------------------------
__global__ void init_maps(int* __restrict__ adj, int* __restrict__ pslot) {
    int t = blockIdx.x * blockDim.x + threadIdx.x;          // 3,145,728 threads
    if (t < G_ * N_ * N_) adj[t] = -1;
    else                  pslot[t - G_ * N_ * N_] = -1;
}

// ---- invert edge scatter: adj[g,i,j] = edge id -----------------------------
__global__ void scatter_edges(const int* __restrict__ ei, int* __restrict__ adj) {
    int e = blockIdx.x * blockDim.x + threadIdx.x;          // 65536
    int g = ei[e], i = ei[NEDGE + e], j = ei[2 * NEDGE + e];
    adj[(g << 14) + (i << 7) + j] = e;
}

// ---- invert path scatter: pslot[g,i,j,d] = path id -------------------------
__global__ void scatter_paths(const int* __restrict__ pi, int* __restrict__ pslot) {
    int p = blockIdx.x * blockDim.x + threadIdx.x;          // 1,048,576
    int g = pi[p], i = pi[NPATH + p], j = pi[2 * NPATH + p], d = pi[3 * NPATH + p];
    pslot[((g << 14) + (i << 7) + j) * D_ + d] = p;
}

// ---- edge encoder: ea16[e,h] = f16( edge_attr[e,:] . edge_enc_w[h,:] ) -----
__global__ void edge_enc(const float* __restrict__ ea, const float* __restrict__ w,
                         _Float16* __restrict__ out) {
    int t = blockIdx.x * blockDim.x + threadIdx.x;          // 2,097,152
    int e = t >> 5, h = t & 31;
    float acc = 0.f;
#pragma unroll
    for (int k = 0; k < 16; ++k) acc += ea[e * 16 + k] * w[h * 16 + k];
    out[t] = (_Float16)acc;
}

// ---- pack W[d] (32x32) into this code's WMMA B lane layout (contiguous K) -----------
// Bpack[hf][d][lane][t] = f16( W[d][16*(lane>>4)+t][16*hf + (lane&15)] )
__global__ void pack_b(const float* __restrict__ edis, _Float16* __restrict__ bp) {
    int t  = blockIdx.x * blockDim.x + threadIdx.x;         // 5120
    int ti = t & 15;
    int u  = t >> 4;
    int lane = u & 31;
    int v  = u >> 5;               // 0..9
    int d  = v % D_;
    int hf = v / D_;
    int k  = 16 * (lane >> 4) + ti;
    int h  = 16 * hf + (lane & 15);
    bp[t] = (_Float16)edis[(d * H_ + k) * H_ + h];
}

// ---- borders: row 0 and column 0 ------------------------------------------
__global__ void border(const float* __restrict__ ab, const float* __restrict__ virt,
                       float* __restrict__ out) {
    int t = blockIdx.x * blockDim.x + threadIdx.x;          // 32*32*257 = 263168
    int g = t / (H_ * 257);
    int r = t % (H_ * 257);
    int h = r / 257;
    int u = r % 257;
    int i = (u < N1) ? 0 : (u - 128);
    int j = (u < N1) ? u : 0;
    float val = 2.0f * ab[(g * N1 + i) * N1 + j] + virt[h];
    out[(((size_t)g * H_ + h) * N1 + i) * N1 + j] = val;
}

// ---- interior: per-distance head mixing via v_wmma_f32_16x16x32_f16 --------
// One wave per 16-pixel tile (contiguous jj). K=32 per distance -> 2 WMMAs/d.
__global__ void __launch_bounds__(256)
mix_interior(const float* __restrict__ ab, const int* __restrict__ spos,
             const int* __restrict__ pidx, const float* __restrict__ spemb,
             const _Float16* __restrict__ ea16, const _Float16* __restrict__ bpack,
             const int* __restrict__ adj, const int* __restrict__ pslot,
             float* __restrict__ out) {
    __shared__ float lds[8][512];                           // 16 KiB / block
    int lane = threadIdx.x & 31;
    int wib  = threadIdx.x >> 5;
    int wave = blockIdx.x * 8 + wib;                        // 32768 waves
    int pix0 = wave * 16;                                   // flattened (g,ii,jj0)
    int g   = pix0 >> 14;
    int rem = pix0 & 16383;
    int ii  = rem >> 7;
    int jj0 = rem & 127;

    int m = lane & 15;                                      // A row (pixel)
    int s = lane >> 4;                                      // K half selector
    const int* myslot = pslot + (pix0 + m) * D_;
    const v16h* Bp = (const v16h*)bpack;

    v8f c0 = {};                                            // heads 0..15
    v8f c1 = {};                                            // heads 16..31
#pragma unroll
    for (int d = 0; d < D_; ++d) {
        // A operand: contiguous K order (same as Bpack) -> lane(m,s) holds K in {16s..16s+15}
        v16h a;
#pragma unroll
        for (int t = 0; t < 16; ++t) a[t] = (_Float16)0.0f;
        int p = myslot[d];
        if (p >= 0) {
            int src = pidx[4 * NPATH + p];
            int dst = pidx[5 * NPATH + p];
            int e = adj[(g << 14) + (src << 7) + dst];
            if (e >= 0) {
                const v8h* bp8 = (const v8h*)(ea16 + (size_t)e * 32);
                v8h lo = bp8[2 * s];
                v8h hi = bp8[2 * s + 1];
#pragma unroll
                for (int t = 0; t < 8; ++t) { a[t] = lo[t]; a[8 + t] = hi[t]; }
            }
        }
        v16h b0 = Bp[(0 * D_ + d) * 32 + lane];
        v16h b1 = Bp[(1 * D_ + d) * 32 + lane];
        c0 = __builtin_amdgcn_wmma_f32_16x16x32_f16(false, a, false, b0, (short)0, c0, false, false);
        c1 = __builtin_amdgcn_wmma_f32_16x16x32_f16(false, a, false, b1, (short)0, c1, false, false);
    }

    // transpose C through LDS: lds[head][pixel]
    int h0 = lane & 15;
    int Mb = (lane >> 4) * 8;
    float* L = lds[wib];
#pragma unroll
    for (int r = 0; r < 8; ++r) {
        L[h0 * 16 + Mb + r]        = c0[r];
        L[(h0 + 16) * 16 + Mb + r] = c1[r];
    }
    __syncthreads();

    // coalesced epilogue: half-wave = one head x 16 contiguous jj
    int jl = lane & 15;
    int jj = jj0 + jl;
    int spv = spos[(g << 14) + (ii << 7) + jj];
    int spc = (spv == 0) ? 1 : spv;
    spc = (spc > 1) ? spc - 1 : spc;
    spc = (spc > D_) ? D_ : spc;
    float inv = 1.0f / (float)spc;
    float ab2 = 2.0f * ab[(g * N1 + ii + 1) * N1 + jj + 1];
#pragma unroll
    for (int q = 0; q < 16; ++q) {
        int head = 2 * q + (lane >> 4);
        float val = L[head * 16 + jl] * inv + ab2 + spemb[spv * H_ + head];
        out[(((size_t)g * H_ + head) * N1 + ii + 1) * N1 + jj + 1] = val;
    }
}

extern "C" void kernel_launch(void* const* d_in, const int* in_sizes, int n_in,
                              void* d_out, int out_size, void* d_ws, size_t ws_size,
                              hipStream_t stream) {
    const float* attn_bias   = (const float*)d_in[0];
    const int*   spatial_pos = (const int*)d_in[1];
    const int*   edge_index  = (const int*)d_in[2];
    const float* edge_attr   = (const float*)d_in[3];
    const int*   path_index  = (const int*)d_in[4];
    const float* edge_enc_w  = (const float*)d_in[5];
    const float* sp_emb      = (const float*)d_in[6];
    const float* virt        = (const float*)d_in[7];
    const float* edge_dis    = (const float*)d_in[8];
    (void)in_sizes; (void)n_in; (void)out_size; (void)ws_size;

    float* out = (float*)d_out;
    char* ws = (char*)d_ws;
    _Float16* bpack = (_Float16*)(ws + OFF_BPACK);
    _Float16* ea16  = (_Float16*)(ws + OFF_EA16);
    int* adj   = (int*)(ws + OFF_ADJ);
    int* pslot = (int*)(ws + OFF_PSLOT);

    init_maps    <<<12288, 256, 0, stream>>>(adj, pslot);
    scatter_edges<<<  256, 256, 0, stream>>>(edge_index, adj);
    scatter_paths<<< 4096, 256, 0, stream>>>(path_index, pslot);
    edge_enc     <<< 8192, 256, 0, stream>>>(edge_attr, edge_enc_w, ea16);
    pack_b       <<<   20, 256, 0, stream>>>(edge_dis, bpack);
    border       <<< 1028, 256, 0, stream>>>(attn_bias, virt, out);
    mix_interior <<< 4096, 256, 0, stream>>>(attn_bias, spatial_pos, path_index,
                                             sp_emb, ea16, bpack, adj, pslot, out);
}
